// CausalSelfAttention_4002909520387
// MI455X (gfx1250) — matrix operand order repaired and checked by reading, not compiled, not run on hardware
//
#include <hip/hip_runtime.h>
#include <hip/hip_bf16.h>
#include <math.h>

// CDNA5 / gfx1250 wave32 WMMA types
typedef __attribute__((ext_vector_type(16))) __bf16 v16bf;
typedef __attribute__((ext_vector_type(8)))  __bf16 v8bf;
typedef __attribute__((ext_vector_type(8)))  float  v8f;
typedef unsigned int u32x4 __attribute__((ext_vector_type(4)));
typedef int          i32x8 __attribute__((ext_vector_type(8)));
typedef int          i32x4 __attribute__((ext_vector_type(4)));

// Tensor Data Mover available? (this toolchain: 6-arg builtin, clang-23 form)
#if defined(__gfx1250__) && defined(__has_builtin)
#if __has_builtin(__builtin_amdgcn_tensor_load_to_lds) && __has_builtin(__builtin_amdgcn_s_wait_tensorcnt)
#define USE_TDM 1
#endif
#endif
#ifndef USE_TDM
#define USE_TDM 0
#endif

union ABFrag { v16bf v; v8bf h[2]; };

static __device__ __forceinline__ v8f wmma_bf16(const ABFrag& a, const ABFrag& b, v8f c) {
    // D = A(16x32 bf16) * B(32x16 bf16) + C(16x16 f32)
    return __builtin_amdgcn_wmma_f32_16x16x32_bf16(
        false, a.v, false, b.v, (short)0, c, false, false);
}

#if USE_TDM
// Issue a 2-D TDM tile load (bf16 elements): tile (tile1 rows x tile0 cols)
// from a row-major tensor with row stride `stride0` (elements) into LDS.
// D# layout per CDNA5 ISA.
static __device__ __forceinline__ void tdm_load_2d(unsigned lds_addr, const void* gptr,
                                                   unsigned tdim0, unsigned tdim1,
                                                   unsigned tile0, unsigned tile1,
                                                   unsigned stride0) {
    unsigned long long ga = (unsigned long long)gptr;
    u32x4 g0;
    g0.x = 1u;                                            // count=1, user descriptor
    g0.y = (unsigned)__builtin_amdgcn_readfirstlane((int)lds_addr);
    g0.z = (unsigned)__builtin_amdgcn_readfirstlane((int)(unsigned)ga);
    g0.w = (unsigned)__builtin_amdgcn_readfirstlane((int)((unsigned)(ga >> 32) | (2u << 30))); // type=2
    i32x8 g1;
    g1[0] = 0x10000;                                      // data_size=2 bytes
    g1[1] = (int)((tdim0 & 0xffffu) << 16);               // tensor_dim0[15:0]
    g1[2] = (int)((tdim0 >> 16) | ((tdim1 & 0xffffu) << 16));
    g1[3] = (int)((tdim1 >> 16) | (tile0 << 16));         // tile_dim0
    g1[4] = (int)tile1;                                   // tile_dim1 (tile_dim2=0)
    g1[5] = (int)stride0;                                 // tensor_dim0_stride[31:0]
    g1[6] = 0;
    g1[7] = 0;
    i32x4 z4 = {0, 0, 0, 0};
    i32x8 z8 = {0, 0, 0, 0, 0, 0, 0, 0};
    __builtin_amdgcn_tensor_load_to_lds(g0, g1, z4, z4, z8, 0);
}
#endif

// ---------------------------------------------------------------------------
// Kernel 1: fp32 -> bf16 elementwise convert
// ---------------------------------------------------------------------------
__global__ void csa_f32_to_bf16(const float* __restrict__ src,
                                __bf16* __restrict__ dst, int n) {
    int i = blockIdx.x * blockDim.x + threadIdx.x;
    if (i < n) dst[i] = (__bf16)src[i];
}

// ---------------------------------------------------------------------------
// Kernel 2: WMMA GEMM  out[m,n] = sum_k A[m,k]*Bw[n,k] + bias[n]
// All dims compile-time (M=8192, N=K=1024) so epilogue index math is shifts.
// One wave computes a 64x64 tile (4x4 WMMA subtiles): 1 vmem per wmma.
// MODE 0: fp32 out row-major;  1: bf16 [B,H,T,D];  2: bf16 [B,H,D,T].
// ---------------------------------------------------------------------------
template <int MODE>
__global__ __launch_bounds__(256)
void csa_gemm_wmma(const __bf16* __restrict__ A, const __bf16* __restrict__ Bw,
                   const float* __restrict__ bias,
                   float* __restrict__ outF, __bf16* __restrict__ outH)
{
    constexpr int M = 8192, N = 1024, K = 1024;
    constexpr int T = 2048, H = 16, D = 64;
    (void)M;

    const int wave = blockIdx.x * (blockDim.x >> 5) + (threadIdx.x >> 5);
    const int tm = wave >> 4;                  // N/64 = 16 tiles wide
    const int tn = wave & 15;

    const int lane = threadIdx.x & 31;
    const int g  = lane >> 4;
    const int ln = lane & 15;

    const int m0 = tm << 6, n0 = tn << 6;
    const __bf16* Abase = A  + (size_t)(m0 + ln) * K;
    const __bf16* Bbase = Bw + (size_t)(n0 + ln) * K;

    v8f acc[4][4];
#pragma unroll
    for (int i = 0; i < 4; ++i)
#pragma unroll
        for (int j = 0; j < 4; ++j)
            acc[i][j] = (v8f){0.f,0.f,0.f,0.f,0.f,0.f,0.f,0.f};

    for (int kk = 0; kk < K; kk += 32) {
        ABFrag a[4], b[4];
#pragma unroll
        for (int i = 0; i < 4; ++i) {
            const __bf16* Ar = Abase + (size_t)(i * 16) * K + kk;
            a[i].h[0] = *(const v8bf*)(Ar + g * 8);        // A: K = g*8..+7
            a[i].h[1] = *(const v8bf*)(Ar + 16 + g * 8);   //    K = 16+g*8..+7
        }
#pragma unroll
        for (int j = 0; j < 4; ++j) {
            const __bf16* Br = Bbase + (size_t)(j * 16) * K + kk;
            b[j].h[0] = *(const v8bf*)(Br + g * 8);        // B: K = g*8..+7
            b[j].h[1] = *(const v8bf*)(Br + 16 + g * 8);
        }
        if (kk + 32 < K) {
            __builtin_prefetch(Abase + kk + 32, 0, 1);
            __builtin_prefetch(Bbase + kk + 32, 0, 1);
        }
#pragma unroll
        for (int i = 0; i < 4; ++i)
#pragma unroll
            for (int j = 0; j < 4; ++j)
                acc[i][j] = wmma_bf16(a[i], b[j], acc[i][j]);
    }

#pragma unroll
    for (int j = 0; j < 4; ++j) {
        const int col = n0 + j * 16 + ln;
        const float bn = bias ? bias[col] : 0.f;
#pragma unroll
        for (int i = 0; i < 4; ++i)
#pragma unroll
            for (int v = 0; v < 8; ++v) {
                const int row = m0 + i * 16 + g * 8 + v;   // C/D: M = g*8+v
                const float val = acc[i][j][v] + bn;
                if (MODE == 0) {
                    outF[(size_t)row * N + col] = val;
                } else {
                    const int b_ = row >> 11, t = row & (T - 1);   // /T, %T
                    const int h  = col >> 6,  d = col & (D - 1);   // /D, %D
                    const size_t idx = (MODE == 1)
                        ? ((((size_t)b_ * H + h) * T + t) * D + d)   // [B,H,T,D]
                        : ((((size_t)b_ * H + h) * D + d) * T + t);  // [B,H,D,T]
                    outH[idx] = (__bf16)val;
                }
            }
    }
}

// ---------------------------------------------------------------------------
// Kernel 3: fused causal flash attention, D=64, bf16 WMMA, f32 state.
// One wave per (b, h, 16-query tile); 2 waves per block.
// K/V tiles are DMA'd into LDS by the Tensor Data Mover (double-buffered,
// synchronized with s_wait_tensorcnt); falls back to direct global loads.
// Q,K: [B,H,T,64] bf16.  Vt: [B,H,64,T] bf16.  Y: [B,T,C] bf16 row-major.
// ---------------------------------------------------------------------------
__global__ __launch_bounds__(64)
void csa_flash_attn_wmma(const __bf16* __restrict__ Q,
                         const __bf16* __restrict__ Km,
                         const __bf16* __restrict__ Vt,
                         __bf16* __restrict__ Y)
{
    constexpr int T = 2048, H = 16, Dh = 64, C = H * Dh;

    __shared__ alignas(128) __bf16 Plds[2][16][32];     // P staging (C->A layout)
#if USE_TDM
    __shared__ alignas(128) __bf16 Kbuf[2][2][32][64];  // [wave][buf][s][d] 16KB
    __shared__ alignas(128) __bf16 Vbuf[2][2][64][32];  // [wave][buf][d][s] 16KB
#endif

    const int wid  = threadIdx.x >> 5;
    const int lane = threadIdx.x & 31;
    const int g  = lane >> 4;
    const int ln = lane & 15;

    const int qt = blockIdx.x * 2 + wid;       // query-tile id
    const int bh = qt >> 7;                    // / (T/16)
    const int t0 = (qt & 127) << 4;

    const __bf16* Qbh = Q  + (size_t)bh * T * Dh;
    const __bf16* Kbh = Km + (size_t)bh * T * Dh;
    const __bf16* Vbh = Vt + (size_t)bh * Dh * T;

    // Q A-fragments, loaded once (row m = ln).
    ABFrag qa[2];
    const __bf16* Qrow = Qbh + (size_t)(t0 + ln) * Dh;
#pragma unroll
    for (int i = 0; i < 2; ++i) {
        const int kk = i * 32;
        qa[i].h[0] = *(const v8bf*)(Qrow + kk + g * 8);
        qa[i].h[1] = *(const v8bf*)(Qrow + kk + 16 + g * 8);
    }

    v8f o[4];
#pragma unroll
    for (int jt = 0; jt < 4; ++jt) o[jt] = (v8f){0.f,0.f,0.f,0.f,0.f,0.f,0.f,0.f};
    float mrow[8], lrow[8];
#pragma unroll
    for (int v = 0; v < 8; ++v) { mrow[v] = -1e30f; lrow[v] = 0.f; }

    const float scale = 0.125f;                // 1/sqrt(64)
    const int sEnd = t0 + 16;                  // causal: keys <= t0+15

#if USE_TDM
    // Preload key/value block 0: K tile 32 rows x 64 cols; V tile 64 rows x 32.
    tdm_load_2d((unsigned)(size_t)&Kbuf[wid][0][0][0], Kbh, 64, T, 64, 32, 64);
    tdm_load_2d((unsigned)(size_t)&Vbuf[wid][0][0][0], Vbh, T, 64, 32, 64, T);
#endif

    int ib = 0;
    for (int s0 = 0; s0 < sEnd; s0 += 32, ++ib) {
        const int pb = ib & 1;
#if USE_TDM
        if (s0 + 32 < sEnd) {   // DMA next block into the other buffer
            tdm_load_2d((unsigned)(size_t)&Kbuf[wid][pb ^ 1][0][0],
                        Kbh + (size_t)(s0 + 32) * Dh, 64, T, 64, 32, 64);
            tdm_load_2d((unsigned)(size_t)&Vbuf[wid][pb ^ 1][0][0],
                        Vbh + (s0 + 32), T, 64, 32, 64, T);
            __builtin_amdgcn_s_wait_tensorcnt(2);   // current block done
        } else {
            __builtin_amdgcn_s_wait_tensorcnt(0);
        }
        asm volatile("" ::: "memory");   // LDS now holds current K/V tiles
#endif
        // ---- S = Q @ K^T for 32 keys as two 16x16 tiles ----
        v8f s_acc[2];
#pragma unroll
        for (int j = 0; j < 2; ++j) {
            s_acc[j] = (v8f){0.f,0.f,0.f,0.f,0.f,0.f,0.f,0.f};
#if USE_TDM
            const __bf16* Krow = &Kbuf[wid][pb][j * 16 + ln][0];
#else
            const __bf16* Krow = Kbh + (size_t)(s0 + j * 16 + ln) * Dh;
#endif
#pragma unroll
            for (int i = 0; i < 2; ++i) {
                const int kk = i * 32;
                ABFrag kb;  // B-fragment: column s=ln, K-contiguous in d
                kb.h[0] = *(const v8bf*)(Krow + kk + g * 8);
                kb.h[1] = *(const v8bf*)(Krow + kk + 16 + g * 8);
                s_acc[j] = wmma_bf16(qa[i], kb, s_acc[j]);
            }
        }

        // ---- scale + causal mask ----
#pragma unroll
        for (int j = 0; j < 2; ++j)
#pragma unroll
            for (int v = 0; v < 8; ++v) {
                float x = s_acc[j][v] * scale;
                const int srow = t0 + g * 8 + v;
                const int scol = s0 + j * 16 + ln;
                s_acc[j][v] = (scol > srow) ? -1e30f : x;
            }

        // ---- online softmax (rows span the 16 lanes of a half-wave) ----
        float alpha[8];
#pragma unroll
        for (int v = 0; v < 8; ++v) {
            float mx = fmaxf(s_acc[0][v], s_acc[1][v]);
#pragma unroll
            for (int off = 1; off < 16; off <<= 1)
                mx = fmaxf(mx, __shfl_xor(mx, off, 32));
            const float mnew = fmaxf(mrow[v], mx);
            alpha[v] = __expf(mrow[v] - mnew);
            mrow[v] = mnew;
        }
#pragma unroll
        for (int v = 0; v < 8; ++v) {
            const float p0 = __expf(s_acc[0][v] - mrow[v]);
            const float p1 = __expf(s_acc[1][v] - mrow[v]);
            s_acc[0][v] = p0;
            s_acc[1][v] = p1;
            float rs = p0 + p1;
#pragma unroll
            for (int off = 1; off < 16; off <<= 1)
                rs += __shfl_xor(rs, off, 32);
            lrow[v] = lrow[v] * alpha[v] + rs;
#pragma unroll
            for (int jt = 0; jt < 4; ++jt) o[jt][v] *= alpha[v];
        }

        // ---- P: C-layout -> A-layout via LDS (same-wave DS ops, in order) ----
#pragma unroll
        for (int j = 0; j < 2; ++j)
#pragma unroll
            for (int v = 0; v < 8; ++v)
                Plds[wid][g * 8 + v][j * 16 + ln] = (__bf16)s_acc[j][v];
        asm volatile("s_wait_dscnt 0" ::: "memory");
        ABFrag pa;  // A-fragment of P (16 rows x 32 keys), row m = ln
        pa.h[0] = *(const v8bf*)(&Plds[wid][ln][g * 8]);
        pa.h[1] = *(const v8bf*)(&Plds[wid][ln][16 + g * 8]);

        // ---- O += P @ V ----
#pragma unroll
        for (int jt = 0; jt < 4; ++jt) {
#if USE_TDM
            const __bf16* Vrow = &Vbuf[wid][pb][jt * 16 + ln][0];
            ABFrag vb;
            vb.h[0] = *(const v8bf*)(Vrow + g * 8);
            vb.h[1] = *(const v8bf*)(Vrow + 16 + g * 8);
#else
            const __bf16* Vrow = Vbh + (size_t)(jt * 16 + ln) * T + s0;
            ABFrag vb;
            vb.h[0] = *(const v8bf*)(Vrow + g * 8);
            vb.h[1] = *(const v8bf*)(Vrow + 16 + g * 8);
#endif
            o[jt] = wmma_bf16(pa, vb, o[jt]);
        }
    }

    // ---- epilogue: normalize, store y at [b, t, h*64 + d] (bf16) ----
    const int b = bh >> 4, h = bh & 15;
#pragma unroll
    for (int jt = 0; jt < 4; ++jt)
#pragma unroll
        for (int v = 0; v < 8; ++v) {
            const float val = o[jt][v] / lrow[v];
            const size_t row = (size_t)b * T + t0 + g * 8 + v;
            const size_t col = (size_t)h * Dh + jt * 16 + ln;
            Y[row * C + col] = (__bf16)val;
        }
}

// ---------------------------------------------------------------------------
// Host orchestration. Inputs: x, Wq, bq, Wk, bk, Wv, bv, Wp, bp (fp32).
// Workspace (256B-aligned chunks), ~92.3 MB total.
// ---------------------------------------------------------------------------
extern "C" void kernel_launch(void* const* d_in, const int* in_sizes, int n_in,
                              void* d_out, int out_size, void* d_ws, size_t ws_size,
                              hipStream_t stream) {
    const float* x  = (const float*)d_in[0];
    const float* Wq = (const float*)d_in[1];
    const float* bq = (const float*)d_in[2];
    const float* Wk = (const float*)d_in[3];
    const float* bk = (const float*)d_in[4];
    const float* Wv = (const float*)d_in[5];
    const float* bv = (const float*)d_in[6];
    const float* Wp = (const float*)d_in[7];
    const float* bp = (const float*)d_in[8];

    const int B = 4, T = 2048, C = 1024;
    const int M = B * T;  // 8192 tokens

    size_t off = 0;
    auto alloc = [&](size_t bytes) -> void* {
        void* p = (char*)d_ws + off;
        off += (bytes + 255) & ~(size_t)255;
        return p;
    };
    __bf16* xb  = (__bf16*)alloc((size_t)M * C * 2);
    __bf16* wqb = (__bf16*)alloc((size_t)C * C * 2);
    __bf16* wkb = (__bf16*)alloc((size_t)C * C * 2);
    __bf16* wvb = (__bf16*)alloc((size_t)C * C * 2);
    __bf16* wpb = (__bf16*)alloc((size_t)C * C * 2);
    __bf16* qb  = (__bf16*)alloc((size_t)M * C * 2);  // [B,H,T,D]
    __bf16* kb  = (__bf16*)alloc((size_t)M * C * 2);  // [B,H,T,D]
    __bf16* vtb = (__bf16*)alloc((size_t)M * C * 2);  // [B,H,D,T]
    __bf16* yb  = (__bf16*)alloc((size_t)M * C * 2);  // [B,T,C]

    // 1) converts
    {
        const int n1 = M * C, n2 = C * C, thr = 256;
        csa_f32_to_bf16<<<(n1 + thr - 1) / thr, thr, 0, stream>>>(x,  xb,  n1);
        csa_f32_to_bf16<<<(n2 + thr - 1) / thr, thr, 0, stream>>>(Wq, wqb, n2);
        csa_f32_to_bf16<<<(n2 + thr - 1) / thr, thr, 0, stream>>>(Wk, wkb, n2);
        csa_f32_to_bf16<<<(n2 + thr - 1) / thr, thr, 0, stream>>>(Wv, wvb, n2);
        csa_f32_to_bf16<<<(n2 + thr - 1) / thr, thr, 0, stream>>>(Wp, wpb, n2);
    }

    // 2) q/k/v projections: 64x64 tile per wave, 8 waves per block
    const int gemmBlocks = ((M >> 6) * (C >> 6)) / 8;   // 256
    csa_gemm_wmma<1><<<gemmBlocks, 256, 0, stream>>>(xb, wqb, bq, nullptr, qb);
    csa_gemm_wmma<1><<<gemmBlocks, 256, 0, stream>>>(xb, wkb, bk, nullptr, kb);
    csa_gemm_wmma<2><<<gemmBlocks, 256, 0, stream>>>(xb, wvb, bv, nullptr, vtb);

    // 3) fused causal flash attention (TDM double-buffered K/V in LDS)
    const int qTiles = 4 * 16 * (T >> 4);               // 8192
    csa_flash_attn_wmma<<<qTiles / 2, 64, 0, stream>>>(qb, kb, vtb, yb);

    // 4) output projection -> fp32 d_out
    csa_gemm_wmma<0><<<gemmBlocks, 256, 0, stream>>>(yb, wpb, bp, (float*)d_out,
                                                     nullptr);
}
